// hlmg_8031588843586
// MI455X (gfx1250) — compile-verified, matrix operand order checked by reading, not run on hardware
//
#include <hip/hip_runtime.h>
#include <hip/hip_bf16.h>
#include <cstdint>
#include <cstddef>

// ---------------------------------------------------------------------------
// Problem dimensions (from reference)
// ---------------------------------------------------------------------------
#define NC    4096      // cells
#define NDG   2048      // drugs
#define DREAL 2040      // feature dim
#define DP    2048      // padded feature dim (multiple of 32 for WMMA K/N)
#define GAMMA_C 15.0f

// ---------------------------------------------------------------------------
// CDNA5 WMMA types
// ---------------------------------------------------------------------------
typedef __bf16 v16bf __attribute__((ext_vector_type(16)));
typedef float  v8f   __attribute__((ext_vector_type(8)));

union Frag16 { v16bf v; unsigned int u[8]; };

__device__ __forceinline__ unsigned short f2bf(float f) {
  unsigned int u = __float_as_uint(f);
  u += 0x7fffu + ((u >> 16) & 1u);        // round-to-nearest-even
  return (unsigned short)(u >> 16);
}

__device__ __forceinline__ float block_sum(float v) {
  __shared__ float red[256];
  int t = threadIdx.x;
  red[t] = v; __syncthreads();
  for (int s = 128; s > 0; s >>= 1) {
    if (t < s) red[t] += red[t + s];
    __syncthreads();
  }
  float r = red[0]; __syncthreads();
  return r;
}

// ---------------------------------------------------------------------------
// CDNA5 async global->LDS copy (ASYNCcnt-tracked), with compile-safe fallback
// ---------------------------------------------------------------------------
#if defined(__HIP_DEVICE_COMPILE__) && __has_builtin(__builtin_amdgcn_global_load_async_to_lds_b64)
#define GEMM_ASYNC 1
#else
#define GEMM_ASYNC 0
#endif

__device__ __forceinline__ void wait_async0() {
#if defined(__HIP_DEVICE_COMPILE__)
#if __has_builtin(__builtin_amdgcn_s_wait_asynccnt)
  __builtin_amdgcn_s_wait_asynccnt(0);
#else
  asm volatile("s_wait_asynccnt 0x0" ::: "memory");
#endif
#endif
}

#if GEMM_ASYNC
typedef int v2i __attribute__((ext_vector_type(2)));
typedef __attribute__((address_space(1))) v2i* v2i_gptr;   // global (AS1) int2*
typedef __attribute__((address_space(3))) v2i* v2i_lptr;   // LDS (AS3) int2*

__device__ __forceinline__ void async_copy_b64(const void* g, void* l) {
  // flat LDS address low 32 bits == LDS offset (ISA aperture mapping)
  __builtin_amdgcn_global_load_async_to_lds_b64(
      (v2i_gptr)(uintptr_t)g, (v2i_lptr)(uintptr_t)l, 0, 0);
}

__device__ __forceinline__ void stage_tiles_async(
    const unsigned short* __restrict__ A, int lda,
    const unsigned short* __restrict__ B, int ldb,
    int m0, int n0, int k0, int sm, int sk,
    unsigned short* sAbuf, unsigned short* sBbuf) {
#pragma unroll
  for (int i = 0; i < 4; ++i) {
    int mm = i * 32 + sm;
    async_copy_b64(A + (size_t)(m0 + mm) * lda + (size_t)(k0 + sk), sAbuf + mm * 32 + sk);
  }
#pragma unroll
  for (int i = 0; i < 4; ++i) {
    int nn = i * 32 + sm;
    async_copy_b64(B + (size_t)(n0 + nn) * ldb + (size_t)(k0 + sk), sBbuf + nn * 32 + sk);
  }
}
#endif

__device__ __forceinline__ void load_tiles_reg(
    const unsigned short* __restrict__ A, int lda,
    const unsigned short* __restrict__ B, int ldb,
    int m0, int n0, int k0, int sm, int sk, uint2* ra, uint2* rb) {
#pragma unroll
  for (int i = 0; i < 4; ++i) {
    int mm = i * 32 + sm;
    ra[i] = *(const uint2*)(A + (size_t)(m0 + mm) * lda + (size_t)(k0 + sk));
    rb[i] = *(const uint2*)(B + (size_t)(n0 + mm) * ldb + (size_t)(k0 + sk));
  }
}

__device__ __forceinline__ void store_tiles_reg(
    const uint2* ra, const uint2* rb, int sm, int sk,
    unsigned short* sAbuf, unsigned short* sBbuf) {
#pragma unroll
  for (int i = 0; i < 4; ++i) {
    int mm = i * 32 + sm;
    *(uint2*)(sAbuf + mm * 32 + sk) = ra[i];
    *(uint2*)(sBbuf + mm * 32 + sk) = rb[i];
  }
}

// ---------------------------------------------------------------------------
// Tiled bf16 WMMA GEMM:  C[M,N] = A[M,K] * B,  B stored K-contiguous [N][K]
//   EPI = 0 : store bf16 row-major C[m*ldc + n]
//   EPI = 1 : relu -> bf16 stored TRANSPOSED C[n*ldc + m]   (ldc = M)
//   EPI = 2 : float C += acc + bias[n] (bias guarded to n_real)
//   EPI = 3 : float C  = sigmoid(gamma * acc / (rn_m[m]*rn_n[n]))
// Tile 128x128x32, 256 threads (8 wave32), 8 WMMA tiles per wave per K step.
// Double-buffered LDS; async global->LDS copies overlapped with WMMA.
// Requires M%128==0, N%128==0, K%32==0 (true at all call sites).
// ---------------------------------------------------------------------------
struct EpiArgs {
  const float* bias;
  const float* rn_m;
  const float* rn_n;
  float gamma;
  int n_real;
};

template <int EPI>
__global__ __launch_bounds__(256)
void gemm_wmma(const unsigned short* __restrict__ A, int lda,
               const unsigned short* __restrict__ B, int ldb,
               void* __restrict__ Cv, int ldc,
               int M, int N, int K, EpiArgs ea) {
  constexpr int BM = 128, BN = 128, BK = 32;
  __shared__ alignas(16) unsigned short sA[2][BM * BK];   // [m][k]
  __shared__ alignas(16) unsigned short sB[2][BN * BK];   // [n][k]

  const int t    = threadIdx.x;
  const int lane = t & 31;
  const int wv   = t >> 5;
  const int wm   = wv >> 2;       // 0..1 : 64-row slab
  const int wn   = wv & 3;        // 0..3 : 32-col slab
  const int h    = lane >> 4;     // K-half selector (ISA 7.12.2)
  const int r16  = lane & 15;
  const int m0   = blockIdx.y * BM;
  const int n0   = blockIdx.x * BN;

  const int eb = t * 4;           // 4 bf16 (8B) per thread per 32-row slice
  const int sm = eb >> 5;         // row within slice
  const int sk = eb & 31;         // k offset (multiple of 4 -> 8B aligned)

  v8f acc[4][2] = {};
  int p = 0;

  // ---- prologue: stage tile 0 ----
#if GEMM_ASYNC
  stage_tiles_async(A, lda, B, ldb, m0, n0, 0, sm, sk, sA[0], sB[0]);
  wait_async0();
#else
  {
    uint2 ra[4], rb[4];
    load_tiles_reg(A, lda, B, ldb, m0, n0, 0, sm, sk, ra, rb);
    store_tiles_reg(ra, rb, sm, sk, sA[0], sB[0]);
  }
#endif
  __syncthreads();

  for (int k0 = 0; k0 < K; k0 += BK) {
    const bool has_next = (k0 + BK) < K;

    // ---- issue copies for next tile into the other buffer ----
#if GEMM_ASYNC
    if (has_next)
      stage_tiles_async(A, lda, B, ldb, m0, n0, k0 + BK, sm, sk, sA[p ^ 1], sB[p ^ 1]);
#else
    uint2 ra[4], rb[4];
    if (has_next)
      load_tiles_reg(A, lda, B, ldb, m0, n0, k0 + BK, sm, sk, ra, rb);
#endif

    // ---- fragments (ISA 7.12.2 16-bit A 16x32 layout; B symmetric) ----
    const unsigned short* sAp = sA[p];
    const unsigned short* sBp = sB[p];
    Frag16 af[4], bfr[2];
#pragma unroll
    for (int mt = 0; mt < 4; ++mt) {
      const unsigned short* row = sAp + (wm * 64 + mt * 16 + r16) * BK;
#pragma unroll
      for (int r = 0; r < 8; ++r) {
        int kb = ((r & 4) ? 16 : 0) + h * 8 + 2 * (r & 3);
        af[mt].u[r] = *(const unsigned int*)(row + kb);
      }
    }
#pragma unroll
    for (int nt = 0; nt < 2; ++nt) {
      const unsigned short* col = sBp + (wn * 32 + nt * 16 + r16) * BK;
#pragma unroll
      for (int r = 0; r < 8; ++r) {
        int kb = ((r & 4) ? 16 : 0) + h * 8 + 2 * (r & 3);
        bfr[nt].u[r] = *(const unsigned int*)(col + kb);
      }
    }

    // ---- 8 WMMAs per wave per K step ----
#pragma unroll
    for (int mt = 0; mt < 4; ++mt)
#pragma unroll
      for (int nt = 0; nt < 2; ++nt)
        acc[mt][nt] = __builtin_amdgcn_wmma_f32_16x16x32_bf16(
            false, af[mt].v, false, bfr[nt].v, (short)0, acc[mt][nt], false, false);

    // ---- retire copies, flip buffers ----
#if GEMM_ASYNC
    if (has_next) wait_async0();
#else
    if (has_next) store_tiles_reg(ra, rb, sm, sk, sA[p ^ 1], sB[p ^ 1]);
#endif
    __syncthreads();
    p ^= 1;
  }

  // ---- epilogue (C/D layout: n = lane&15, m = r + 8*(lane>=16)) ----
#pragma unroll
  for (int mt = 0; mt < 4; ++mt) {
#pragma unroll
    for (int nt = 0; nt < 2; ++nt) {
      int n = n0 + wn * 32 + nt * 16 + r16;
#pragma unroll
      for (int r = 0; r < 8; ++r) {
        int m = m0 + wm * 64 + mt * 16 + r + h * 8;
        float v = acc[mt][nt][r];
        if (EPI == 0) {
          ((unsigned short*)Cv)[(size_t)m * ldc + n] = f2bf(v);
        } else if (EPI == 1) {
          ((unsigned short*)Cv)[(size_t)n * ldc + m] = f2bf(fmaxf(v, 0.f));
        } else if (EPI == 2) {
          float bv = (n < ea.n_real) ? ea.bias[n] : 0.f;
          ((float*)Cv)[(size_t)m * ldc + n] += v + bv;
        } else {
          float den = ea.rn_m[m] * ea.rn_n[n];
          float c = v / den;
          ((float*)Cv)[(size_t)m * ldc + n] = 1.f / (1.f + __expf(-ea.gamma * c));
        }
      }
    }
  }
}

// ---------------------------------------------------------------------------
// Elementwise / reduction kernels
// ---------------------------------------------------------------------------
__global__ void k_fill0(float* p, size_t n) {
  size_t i = (size_t)blockIdx.x * blockDim.x + threadIdx.x;
  size_t stride = (size_t)gridDim.x * blockDim.x;
  for (; i < n; i += stride) p[i] = 0.f;
}

__global__ __launch_bounds__(256)
void k_rowsum(const float* __restrict__ A, int cols, float* __restrict__ out) {
  int row = blockIdx.x;
  const float* r = A + (size_t)row * cols;
  float s = 0.f;
  for (int j = threadIdx.x; j < cols; j += 256) s += r[j];
  float tot = block_sum(s);
  if (threadIdx.x == 0) out[row] = tot;
}

__global__ __launch_bounds__(256)
void k_colsum(const float* __restrict__ A, int rows, int cols, float* __restrict__ out) {
  int j = blockIdx.x * 256 + threadIdx.x;
  if (j >= cols) return;
  float s = 0.f;
  for (int i = 0; i < rows; ++i) s += A[(size_t)i * cols + j];
  out[j] = s;
}

__global__ void k_deg_derive(const float* __restrict__ deg, int n,
                             float* __restrict__ dx, float* __restrict__ selfv) {
  int i = blockIdx.x * 256 + threadIdx.x;
  if (i >= n) return;
  float d = deg[i] + 1.f;
  dx[i] = rsqrtf(d);
  selfv[i] = 1.f / d + 1.f;
}

__global__ void k_pow_nh(const float* __restrict__ a, int n, float* __restrict__ out) {
  int i = blockIdx.x * 256 + threadIdx.x;
  if (i >= n) return;
  out[i] = rsqrtf(a[i]);
}

// per-row 7th-largest: per-thread sorted top-7, then LDS tree-merge
__global__ __launch_bounds__(256)
void k_topk7(const float* __restrict__ sim, int n, float* __restrict__ thr) {
  __shared__ float ls[256 * 7];
  int row = blockIdx.x, t = threadIdx.x;
  const float* r = sim + (size_t)row * n;
  float top[7];
#pragma unroll
  for (int k = 0; k < 7; ++k) top[k] = -3.0e38f;
  for (int j = t; j < n; j += 256) {
    float v = r[j];
    if (v > top[6]) {
      top[6] = v;
#pragma unroll
      for (int k = 6; k > 0; --k)
        if (top[k] > top[k - 1]) { float tmp = top[k]; top[k] = top[k - 1]; top[k - 1] = tmp; }
    }
  }
#pragma unroll
  for (int k = 0; k < 7; ++k) ls[t * 7 + k] = top[k];
  __syncthreads();
  for (int s = 128; s > 0; s >>= 1) {
    if (t < s) {
      float a[7], b[7], o[7];
#pragma unroll
      for (int k = 0; k < 7; ++k) { a[k] = ls[t * 7 + k]; b[k] = ls[(t + s) * 7 + k]; }
      int ia = 0, ib = 0;
#pragma unroll
      for (int k = 0; k < 7; ++k) {
        if (a[ia] >= b[ib]) o[k] = a[ia++]; else o[k] = b[ib++];
      }
#pragma unroll
      for (int k = 0; k < 7; ++k) ls[t * 7 + k] = o[k];
    }
    __syncthreads();
  }
  if (t == 0) thr[row] = ls[6];
}

__global__ __launch_bounds__(256)
void k_knn_rowsum(const float* __restrict__ sim, const float* __restrict__ thr,
                  int n, float* __restrict__ out) {
  int row = blockIdx.x;
  const float* r = sim + (size_t)row * n;
  float th = thr[row];
  float s = 0.f;
  for (int j = threadIdx.x; j < n; j += 256) {
    float v = r[j];
    s += (v >= th) ? v : 0.f;
  }
  float tot = block_sum(s);
  if (threadIdx.x == 0) out[row] = tot;
}

__global__ __launch_bounds__(256)
void k_knn_colsum(const float* __restrict__ sim, const float* __restrict__ thr,
                  int n, float* __restrict__ out) {
  int j = blockIdx.x * 256 + threadIdx.x;
  if (j >= n) return;
  float s = 0.f;
  for (int i = 0; i < n; ++i) {
    float v = sim[(size_t)i * n + j];
    s += (v >= thr[i]) ? v : 0.f;
  }
  out[j] = s;
}

__global__ __launch_bounds__(256)
void k_build_lap(const float* __restrict__ sim, const float* __restrict__ thr,
                 const float* __restrict__ dx, const float* __restrict__ dy,
                 int n, unsigned short* __restrict__ lapb) {
  int i = blockIdx.y;
  int j = blockIdx.x * 256 + threadIdx.x;
  if (j >= n) return;
  float v = sim[(size_t)i * n + j];
  v = (v >= thr[i]) ? v : 0.f;
  lapb[(size_t)i * n + j] = f2bf(v * dx[i] * dy[j]);
}

// row-major padded bf16 (A-side operands)
__global__ __launch_bounds__(256)
void k_build_pad(const float* __restrict__ x, int dreal, int dpad,
                 const float* __restrict__ rowscale, unsigned short* __restrict__ out) {
  int i = blockIdx.y;
  int d = blockIdx.x * 256 + threadIdx.x;
  if (d >= dpad) return;
  float v = 0.f;
  if (d < dreal) {
    v = x[(size_t)i * dreal + d];
    if (rowscale) v *= rowscale[i];
  }
  out[(size_t)i * dpad + d] = f2bf(v);
}

// LDS-tiled transpose + pad + f32->bf16:  out[d][i] = x[i][d]  (out is [DP][rows])
__global__ __launch_bounds__(256)
void k_transpose_pad(const float* __restrict__ x, int rows, int dreal,
                     unsigned short* __restrict__ out) {
  __shared__ float tile[64][65];
  int tx = threadIdx.x & 63;
  int ty = threadIdx.x >> 6;          // 0..3
  int i0 = blockIdx.x * 64;
  int d0 = blockIdx.y * 64;
#pragma unroll
  for (int r = 0; r < 16; ++r) {
    int i = i0 + ty + r * 4;
    int d = d0 + tx;
    float v = (d < dreal) ? x[(size_t)i * dreal + d] : 0.f;
    tile[ty + r * 4][tx] = v;
  }
  __syncthreads();
#pragma unroll
  for (int r = 0; r < 16; ++r) {
    int dl = ty + r * 4;
    out[(size_t)(d0 + dl) * rows + (i0 + tx)] = f2bf(tile[tx][dl]);
  }
}

__global__ __launch_bounds__(256)
void k_build_agg(const float* __restrict__ adj, const float* __restrict__ dxc,
                 const float* __restrict__ dyd,
                 unsigned short* __restrict__ aggc, unsigned short* __restrict__ aggd) {
  int i = blockIdx.y;                        // cell
  int j = blockIdx.x * 256 + threadIdx.x;    // drug
  if (j >= NDG) return;
  float v = adj[(size_t)i * NDG + j] * dxc[i] * dyd[j];
  unsigned short b = f2bf(v);
  aggc[(size_t)i * NDG + j] = b;
  aggd[(size_t)j * NC + i] = b;              // agg_drug == agg_cell^T
}

__global__ __launch_bounds__(256)
void k_build_w(const float* __restrict__ W, int dreal, int dpad,
               unsigned short* __restrict__ out) {
  int o = blockIdx.y;
  int k = blockIdx.x * 256 + threadIdx.x;
  if (k >= dpad) return;
  float v = (o < dreal && k < dreal) ? W[(size_t)o * dreal + k] : 0.f;
  out[(size_t)o * dpad + k] = f2bf(v);
}

// relu(S*(1+gate)), row-center, row-norm; writes centered bf16 (pad cols = 0)
__global__ __launch_bounds__(256)
void k_rowfin(const float* __restrict__ S, const float* __restrict__ gate,
              int dreal, int dpad, unsigned short* __restrict__ xb,
              float* __restrict__ rn) {
  int row = blockIdx.x, t = threadIdx.x;
  float vals[8];
  float s = 0.f;
#pragma unroll
  for (int c = 0; c < 8; ++c) {
    int col = c * 256 + t;
    float v = 0.f;
    if (col < dreal) {
      float g = gate[(size_t)row * dreal + col];
      v = fmaxf(S[(size_t)row * dpad + col] * (1.f + g), 0.f);
    }
    vals[c] = v; s += v;
  }
  float mean = block_sum(s) / (float)dreal;
  float ss = 0.f;
#pragma unroll
  for (int c = 0; c < 8; ++c) {
    int col = c * 256 + t;
    float cv = (col < dreal) ? (vals[c] - mean) : 0.f;
    xb[(size_t)row * dpad + col] = f2bf(cv);
    ss += cv * cv;
  }
  float tot = block_sum(ss);
  if (t == 0) rn[row] = sqrtf(tot);
}

// ---------------------------------------------------------------------------
// Host orchestration
// ---------------------------------------------------------------------------
extern "C" void kernel_launch(void* const* d_in, const int* in_sizes, int n_in,
                              void* d_out, int out_size, void* d_ws, size_t ws_size,
                              hipStream_t stream) {
  (void)in_sizes; (void)n_in; (void)out_size; (void)ws_size;

  const float* adj      = (const float*)d_in[0];
  const float* cell_sim = (const float*)d_in[1];
  const float* drug_sim = (const float*)d_in[2];
  const float* exp_     = (const float*)d_in[3];
  const float* fig      = (const float*)d_in[4];
  const float* W_sc = (const float*)d_in[5];
  const float* W_sd = (const float*)d_in[6];
  const float* W_cs = (const float*)d_in[7];  const float* b_cs = (const float*)d_in[8];
  const float* W_c1 = (const float*)d_in[9];  const float* b_c1 = (const float*)d_in[10];
  const float* W_c2 = (const float*)d_in[11]; const float* b_c2 = (const float*)d_in[12];
  const float* W_ds = (const float*)d_in[13]; const float* b_ds = (const float*)d_in[14];
  const float* W_d1 = (const float*)d_in[15]; const float* b_d1 = (const float*)d_in[16];
  const float* W_d2 = (const float*)d_in[17]; const float* b_d2 = (const float*)d_in[18];

  char* base = (char*)d_ws;
  size_t off = 0;
  auto alloc = [&](size_t bytes) -> void* {
    void* p = base + off;
    off += (bytes + 255) & ~(size_t)255;
    return p;
  };

  // bf16 buffers (B-side operands stored K-contiguous / transposed)
  unsigned short* expT = (unsigned short*)alloc((size_t)DP * NC * 2);    // [DP][NC]
  unsigned short* figT = (unsigned short*)alloc((size_t)DP * NDG * 2);   // [DP][NDG]
  unsigned short* lapb = (unsigned short*)alloc((size_t)NC * NC * 2);    // lap_c / lap_d / agg_c+agg_d
  unsigned short* tbuf = (unsigned short*)alloc((size_t)NC * DP * 2);    // T_ce / T_df
  unsigned short* sfcT = (unsigned short*)alloc((size_t)DP * NC * 2);    // [DP][NC]; later Y1c
  unsigned short* sfdT = (unsigned short*)alloc((size_t)DP * NDG * 2);   // [DP][NDG]; later Y1d
  unsigned short* uc1  = (unsigned short*)alloc((size_t)NC * DP * 2);
  unsigned short* uc2  = (unsigned short*)alloc((size_t)NC * DP * 2);
  unsigned short* ud1  = (unsigned short*)alloc((size_t)NDG * DP * 2);
  unsigned short* ud2  = (unsigned short*)alloc((size_t)NDG * DP * 2);
  unsigned short* wb   = (unsigned short*)alloc((size_t)DP * DP * 2);    // weight staging [out][in]
  float* Sc  = (float*)alloc((size_t)NC * DP * 4);
  float* Sd  = (float*)alloc((size_t)NDG * DP * 4);
  unsigned short* xcb = (unsigned short*)alloc((size_t)NC * DP * 2);
  unsigned short* ycb = (unsigned short*)alloc((size_t)NDG * DP * 2);
  // vectors
  float* deg_c = (float*)alloc(NC * 4);   float* deg_d = (float*)alloc(NDG * 4);
  float* dxc   = (float*)alloc(NC * 4);   float* selfc = (float*)alloc(NC * 4);
  float* dyd   = (float*)alloc(NDG * 4);  float* selfd = (float*)alloc(NDG * 4);
  float* thr_c = (float*)alloc(NC * 4);   float* thr_d = (float*)alloc(NDG * 4);
  float* arow_c = (float*)alloc(NC * 4);  float* acol_c = (float*)alloc(NC * 4);
  float* arow_d = (float*)alloc(NDG * 4); float* acol_d = (float*)alloc(NDG * 4);
  float* dxl_c = (float*)alloc(NC * 4);   float* dyl_c = (float*)alloc(NC * 4);
  float* dxl_d = (float*)alloc(NDG * 4);  float* dyl_d = (float*)alloc(NDG * 4);
  float* rn_c  = (float*)alloc(NC * 4);   float* rn_d  = (float*)alloc(NDG * 4);

  unsigned short* aggc = lapb;                             // 16MB
  unsigned short* aggd = lapb + (size_t)NC * NDG;          // 16MB

  dim3 blk(256);
  EpiArgs ea0{nullptr, nullptr, nullptr, 0.f, 0};

  // -- adjacency degrees / self-loop weights --
  k_rowsum<<<NC, blk, 0, stream>>>(adj, NDG, deg_c);
  k_colsum<<<(NDG + 255) / 256, blk, 0, stream>>>(adj, NC, NDG, deg_d);
  k_deg_derive<<<(NC + 255) / 256, blk, 0, stream>>>(deg_c, NC, dxc, selfc);
  k_deg_derive<<<(NDG + 255) / 256, blk, 0, stream>>>(deg_d, NDG, dyd, selfd);

  // -- kNN thresholds and laplacian degree vectors --
  k_topk7<<<NC, blk, 0, stream>>>(cell_sim, NC, thr_c);
  k_topk7<<<NDG, blk, 0, stream>>>(drug_sim, NDG, thr_d);
  k_knn_rowsum<<<NC, blk, 0, stream>>>(cell_sim, thr_c, NC, arow_c);
  k_knn_colsum<<<NC / 256, blk, 0, stream>>>(cell_sim, thr_c, NC, acol_c);
  k_knn_rowsum<<<NDG, blk, 0, stream>>>(drug_sim, thr_d, NDG, arow_d);
  k_knn_colsum<<<NDG / 256, blk, 0, stream>>>(drug_sim, thr_d, NDG, acol_d);
  k_pow_nh<<<NC / 256, blk, 0, stream>>>(arow_c, NC, dxl_c);
  k_pow_nh<<<NC / 256, blk, 0, stream>>>(acol_c, NC, dyl_c);
  k_pow_nh<<<NDG / 256, blk, 0, stream>>>(arow_d, NDG, dxl_d);
  k_pow_nh<<<NDG / 256, blk, 0, stream>>>(acol_d, NDG, dyl_d);

  // -- transposed bf16 feature matrices (B-side) --
  k_transpose_pad<<<dim3(NC / 64, DP / 64), blk, 0, stream>>>(exp_, NC, DREAL, expT);
  k_transpose_pad<<<dim3(NDG / 64, DP / 64), blk, 0, stream>>>(fig, NDG, DREAL, figT);

  // -- cell SimLayer: sf_c = relu((lap_c @ exp) @ W_sc^T), stored transposed --
  k_build_lap<<<dim3(NC / 256, NC), blk, 0, stream>>>(cell_sim, thr_c, dxl_c, dyl_c, NC, lapb);
  gemm_wmma<0><<<dim3(DP / 128, NC / 128), blk, 0, stream>>>(lapb, NC, expT, NC, tbuf, DP, NC, DP, NC, ea0);
  k_build_w<<<dim3(DP / 256, DP), blk, 0, stream>>>(W_sc, DREAL, DP, wb);
  gemm_wmma<1><<<dim3(DP / 128, NC / 128), blk, 0, stream>>>(tbuf, DP, wb, DP, sfcT, NC, NC, DP, DP, ea0);

  // -- drug SimLayer: sf_d = relu((lap_d @ fig) @ W_sd^T), stored transposed --
  k_build_lap<<<dim3(NDG / 256, NDG), blk, 0, stream>>>(drug_sim, thr_d, dxl_d, dyl_d, NDG, lapb);
  gemm_wmma<0><<<dim3(DP / 128, NDG / 128), blk, 0, stream>>>(lapb, NDG, figT, NDG, tbuf, DP, NDG, DP, NDG, ea0);
  k_build_w<<<dim3(DP / 256, DP), blk, 0, stream>>>(W_sd, DREAL, DP, wb);
  gemm_wmma<1><<<dim3(DP / 128, NDG / 128), blk, 0, stream>>>(tbuf, DP, wb, DP, sfdT, NDG, NDG, DP, DP, ea0);

  // -- aggregation features --
  k_build_agg<<<dim3(NDG / 256, NC), blk, 0, stream>>>(adj, dxc, dyd, aggc, aggd);
  gemm_wmma<0><<<dim3(DP / 128, NC / 128), blk, 0, stream>>>(aggc, NDG, figT, NDG, uc1, DP, NC, DP, NDG, ea0);
  gemm_wmma<0><<<dim3(DP / 128, NC / 128), blk, 0, stream>>>(aggc, NDG, sfdT, NDG, uc2, DP, NC, DP, NDG, ea0);
  gemm_wmma<0><<<dim3(DP / 128, NDG / 128), blk, 0, stream>>>(aggd, NC, expT, NC, ud1, DP, NDG, DP, NC, ea0);
  gemm_wmma<0><<<dim3(DP / 128, NDG / 128), blk, 0, stream>>>(aggd, NC, sfcT, NC, ud2, DP, NDG, DP, NC, ea0);

  // -- self-loop features (reuse sfcT/sfdT, both fully consumed above) --
  unsigned short* y1c = sfcT;
  unsigned short* y1d = sfdT;
  k_build_pad<<<dim3(DP / 256, NC), blk, 0, stream>>>(exp_, DREAL, DP, selfc, y1c);
  k_build_pad<<<dim3(DP / 256, NDG), blk, 0, stream>>>(fig, DREAL, DP, selfd, y1d);

  // -- accumulate the three gated linears per side:  S = sum(X_j @ W_j^T + b_j) --
  k_fill0<<<2048, blk, 0, stream>>>(Sc, (size_t)NC * DP);
  k_fill0<<<2048, blk, 0, stream>>>(Sd, (size_t)NDG * DP);

  EpiArgs eac{nullptr, nullptr, nullptr, 0.f, DREAL};
  eac.bias = b_cs;
  k_build_w<<<dim3(DP / 256, DP), blk, 0, stream>>>(W_cs, DREAL, DP, wb);
  gemm_wmma<2><<<dim3(DP / 128, NC / 128), blk, 0, stream>>>(y1c, DP, wb, DP, Sc, DP, NC, DP, DP, eac);
  eac.bias = b_c1;
  k_build_w<<<dim3(DP / 256, DP), blk, 0, stream>>>(W_c1, DREAL, DP, wb);
  gemm_wmma<2><<<dim3(DP / 128, NC / 128), blk, 0, stream>>>(uc1, DP, wb, DP, Sc, DP, NC, DP, DP, eac);
  eac.bias = b_c2;
  k_build_w<<<dim3(DP / 256, DP), blk, 0, stream>>>(W_c2, DREAL, DP, wb);
  gemm_wmma<2><<<dim3(DP / 128, NC / 128), blk, 0, stream>>>(uc2, DP, wb, DP, Sc, DP, NC, DP, DP, eac);
  eac.bias = b_ds;
  k_build_w<<<dim3(DP / 256, DP), blk, 0, stream>>>(W_ds, DREAL, DP, wb);
  gemm_wmma<2><<<dim3(DP / 128, NDG / 128), blk, 0, stream>>>(y1d, DP, wb, DP, Sd, DP, NDG, DP, DP, eac);
  eac.bias = b_d1;
  k_build_w<<<dim3(DP / 256, DP), blk, 0, stream>>>(W_d1, DREAL, DP, wb);
  gemm_wmma<2><<<dim3(DP / 128, NDG / 128), blk, 0, stream>>>(ud1, DP, wb, DP, Sd, DP, NDG, DP, DP, eac);
  eac.bias = b_d2;
  k_build_w<<<dim3(DP / 256, DP), blk, 0, stream>>>(W_d2, DREAL, DP, wb);
  gemm_wmma<2><<<dim3(DP / 128, NDG / 128), blk, 0, stream>>>(ud2, DP, wb, DP, Sd, DP, NDG, DP, DP, eac);

  // -- relu(S*(1+gate)), center, norm --
  k_rowfin<<<NC, blk, 0, stream>>>(Sc, exp_, DREAL, DP, xcb, rn_c);
  k_rowfin<<<NDG, blk, 0, stream>>>(Sd, fig, DREAL, DP, ycb, rn_d);

  // -- decoder: sigmoid(GAMMA * (xc @ yc^T) / (||xc|| * ||yc||)) --
  EpiArgs eaf{nullptr, rn_c, rn_d, GAMMA_C, DREAL};
  gemm_wmma<3><<<dim3(NDG / 128, NC / 128), blk, 0, stream>>>(xcb, DP, ycb, DP, d_out, NDG, NC, NDG, DP, eaf);
}
